// AgentSelfAttention1d_23390391894153
// MI455X (gfx1250) — matrix operand order repaired and checked by reading, not compiled, not run on hardware
//
#include <hip/hip_runtime.h>

// ---------------------------------------------------------------------------
// AgentSelfAttention1d for MI455X (gfx1250, wave32, WMMA 16x16x32 f16).
// B=8, D=512, L=4096, P=128 agent tokens, SEG=L/P=32.
//
// All matmuls run on v_wmma_f32_16x16x32_f16 (f16 A/B, f32 accumulate).
// Each GEMM wave computes a 32(M)x64(N) tile: 2 A-frags + 4 B-frags per
// k-step -> 8 WMMAs per 12 b128 loads. Intermediates are staged in global
// workspace (~160MB, essentially resident in the 192MB L2).
// ---------------------------------------------------------------------------

typedef __attribute__((ext_vector_type(16))) _Float16 v16h;
typedef __attribute__((ext_vector_type(8)))  float    v8f;

#define BB   8
#define DD   512
#define LL   4096
#define PP   128
#define SEG  32
#define BL   (BB * LL)   // 32768 token rows
#define BP   (BB * PP)   // 1024 agent rows

// ---- WMMA fragment loaders (contiguous k order, same for A and B) -----
// A (16x32 f16, MxK): lane m=lane&15, h=lane>>4. j -> K = j + 16h.
__device__ __forceinline__ v16h load_afrag(const _Float16* __restrict__ row, int h) {
  union { v16h v; uint4 q[2]; } u;
  u.q[0] = *reinterpret_cast<const uint4*>(row + 16 * h);
  u.q[1] = *reinterpret_cast<const uint4*>(row + 16 * h + 8);
  return u.v;
}
// B (32x16 f16, KxN): lane n=lane&15, h=lane>>4. j -> K = j + 16h.
__device__ __forceinline__ v16h load_bfrag(const _Float16* __restrict__ col, int h) {
  union { v16h v; uint4 q[2]; } u;
  const uint4* p = reinterpret_cast<const uint4*>(col + 16 * h);
  u.q[0] = p[0];
  u.q[1] = p[1];
  return u.v;
}

__device__ __forceinline__ v8f wmma_f16(v16h a, v16h b, v8f c) {
  return __builtin_amdgcn_wmma_f32_16x16x32_f16(false, a, false, b, (short)0, c,
                                                false, false);
}

// ds_swizzle XOR lane reduction (wave32): imm = (xor_mask<<10) | 0x1f
template <int IMM>
__device__ __forceinline__ float swzf(float x) {
  int i = __builtin_amdgcn_ds_swizzle(__builtin_bit_cast(int, x), IMM);
  return __builtin_bit_cast(float, i);
}
__device__ __forceinline__ float wave_max(float v) {
  v = fmaxf(v, swzf<0x041F>(v));
  v = fmaxf(v, swzf<0x081F>(v));
  v = fmaxf(v, swzf<0x101F>(v));
  v = fmaxf(v, swzf<0x201F>(v));
  v = fmaxf(v, swzf<0x401F>(v));
  return v;
}
__device__ __forceinline__ float wave_sum(float v) {
  v += swzf<0x041F>(v);
  v += swzf<0x081F>(v);
  v += swzf<0x101F>(v);
  v += swzf<0x201F>(v);
  v += swzf<0x401F>(v);
  return v;
}

// ---- 1) x [B,D,L] f32  ->  xt [B,L,D] f16 (LDS-tiled transpose+convert) ---
__global__ __launch_bounds__(256) void k_transpose_x(const float* __restrict__ x,
                                                     _Float16* __restrict__ xt) {
  __shared__ float t[32][33];
  int tx = threadIdx.x, ty = threadIdx.y;
  int l0 = blockIdx.x * 32, d0 = blockIdx.y * 32, b = blockIdx.z;
  const float* xb = x + (size_t)b * DD * LL;
#pragma unroll
  for (int i = 0; i < 4; ++i)
    t[ty + 8 * i][tx] = xb[(size_t)(d0 + ty + 8 * i) * LL + l0 + tx];
  __syncthreads();
  _Float16* xo = xt + (size_t)b * LL * DD;
#pragma unroll
  for (int i = 0; i < 4; ++i)
    xo[(size_t)(l0 + ty + 8 * i) * DD + d0 + tx] = (_Float16)t[tx][ty + 8 * i];
}

// ---- 2) Wq/Wk/Wv f32 -> f16 ([d_out, d_in] row-major == B-operand layout) -
__global__ __launch_bounds__(256) void k_convert_w(const float* __restrict__ wq,
                                                   const float* __restrict__ wk,
                                                   const float* __restrict__ wv,
                                                   _Float16* __restrict__ w3) {
  int i = blockIdx.x * 256 + threadIdx.x;
  const float* src = (i < DD * DD) ? wq : (i < 2 * DD * DD) ? wk : wv;
  w3[i] = (_Float16)src[i % (DD * DD)];
}

// ---- 3) projections: q/k [BL,D] f16 ; v stored transposed vT [B,D,L] f16 --
__global__ __launch_bounds__(128) void k_proj(const _Float16* __restrict__ xt,
                                              const _Float16* __restrict__ w3,
                                              const float* __restrict__ bq,
                                              const float* __restrict__ bk,
                                              const float* __restrict__ bv,
                                              _Float16* __restrict__ qb,
                                              _Float16* __restrict__ kb,
                                              _Float16* __restrict__ vtb) {
  int lane = threadIdx.x & 31, wave = threadIdx.x >> 5;
  int which = blockIdx.z;
  const _Float16* W = w3 + (size_t)which * DD * DD;
  const float* bias = (which == 0) ? bq : (which == 1) ? bk : bv;
  int m0 = blockIdx.x * 32;
  int n0 = blockIdx.y * 256 + wave * 64;
  int ml = lane & 15, h = lane >> 4;
  const _Float16* arow0 = xt + (size_t)(m0 + ml) * DD;
  const _Float16* arow1 = xt + (size_t)(m0 + 16 + ml) * DD;
  v8f z = {0.f, 0.f, 0.f, 0.f, 0.f, 0.f, 0.f, 0.f};
  v8f acc[2][4] = {{z, z, z, z}, {z, z, z, z}};
  for (int k0 = 0; k0 < DD; k0 += 32) {
    v16h a0 = load_afrag(arow0 + k0, h);
    v16h a1 = load_afrag(arow1 + k0, h);
#pragma unroll
    for (int f = 0; f < 4; ++f) {
      v16h bf = load_bfrag(W + (size_t)(n0 + f * 16 + ml) * DD + k0, h);
      acc[0][f] = wmma_f16(a0, bf, acc[0][f]);
      acc[1][f] = wmma_f16(a1, bf, acc[1][f]);
    }
  }
#pragma unroll
  for (int f = 0; f < 4; ++f) {
    int n = n0 + f * 16 + ml;
    float bsc = bias[n];
#pragma unroll
    for (int t = 0; t < 2; ++t) {
#pragma unroll
      for (int r = 0; r < 8; ++r) {
        int m = m0 + 16 * t + r + 8 * h;
        _Float16 hv = (_Float16)(acc[t][f][r] + bsc);
        if (which == 0)      qb[(size_t)m * DD + n] = hv;
        else if (which == 1) kb[(size_t)m * DD + n] = hv;
        else {
          int b = m >> 12, l = m & (LL - 1);
          vtb[((size_t)b * DD + n) * LL + l] = hv;
        }
      }
    }
  }
}

// ---- 4) agent pooling: a[b,p,d] = mean over SEG rows of q --------------
__global__ __launch_bounds__(256) void k_pool(const _Float16* __restrict__ qb,
                                              _Float16* __restrict__ ab) {
  int g = blockIdx.x;                 // b*P + p
  int b = g >> 7, p = g & (PP - 1);
  const _Float16* base = qb + ((size_t)b * LL + (size_t)p * SEG) * DD;
#pragma unroll
  for (int c = 0; c < 2; ++c) {
    int d = threadIdx.x + 256 * c;
    float s = 0.f;
#pragma unroll 4
    for (int i = 0; i < SEG; ++i) s += (float)base[(size_t)i * DD + d];
    ab[(size_t)g * DD + d] = (_Float16)(s * (1.0f / SEG));
  }
}

// ---- 5) stage-1 scores: S[bP+p, l] = a . k  (K=D) ------------------------
__global__ __launch_bounds__(128) void k_s1(const _Float16* __restrict__ ab,
                                            const _Float16* __restrict__ kb,
                                            float* __restrict__ S) {
  int lane = threadIdx.x & 31, wave = threadIdx.x >> 5;
  int m0 = blockIdx.x * 32;           // over B*P
  int b = m0 >> 7;
  int n0 = blockIdx.y * 256 + wave * 64;
  int ml = lane & 15, h = lane >> 4;
  const _Float16* arow0 = ab + (size_t)(m0 + ml) * DD;
  const _Float16* arow1 = ab + (size_t)(m0 + 16 + ml) * DD;
  v8f z = {0.f, 0.f, 0.f, 0.f, 0.f, 0.f, 0.f, 0.f};
  v8f acc[2][4] = {{z, z, z, z}, {z, z, z, z}};
  for (int k0 = 0; k0 < DD; k0 += 32) {
    v16h a0 = load_afrag(arow0 + k0, h);
    v16h a1 = load_afrag(arow1 + k0, h);
#pragma unroll
    for (int f = 0; f < 4; ++f) {
      v16h bf = load_bfrag(kb + ((size_t)b * LL + n0 + f * 16 + ml) * DD + k0, h);
      acc[0][f] = wmma_f16(a0, bf, acc[0][f]);
      acc[1][f] = wmma_f16(a1, bf, acc[1][f]);
    }
  }
#pragma unroll
  for (int f = 0; f < 4; ++f) {
    int n = n0 + f * 16 + ml;
#pragma unroll
    for (int t = 0; t < 2; ++t)
#pragma unroll
      for (int r = 0; r < 8; ++r)
        S[(size_t)(m0 + 16 * t + r + 8 * h) * LL + n] = acc[t][f][r];
  }
}

// ---- 6) softmax over l (width 4096), write f16 probs ---------------------
__global__ __launch_bounds__(256) void k_softmax1(const float* __restrict__ S,
                                                  _Float16* __restrict__ pr) {
  __shared__ float red[256];
  int row = blockIdx.x, t = threadIdx.x;
  const float* s = S + (size_t)row * LL;
  float mx = -1e30f;
  for (int i = t; i < LL; i += 256) mx = fmaxf(mx, s[i]);
  red[t] = mx; __syncthreads();
  for (int o = 128; o > 0; o >>= 1) {
    if (t < o) red[t] = fmaxf(red[t], red[t + o]);
    __syncthreads();
  }
  mx = red[0]; __syncthreads();
  float sm = 0.f;
  for (int i = t; i < LL; i += 256) sm += __expf(s[i] - mx);
  red[t] = sm; __syncthreads();
  for (int o = 128; o > 0; o >>= 1) {
    if (t < o) red[t] += red[t + o];
    __syncthreads();
  }
  float inv = 1.0f / red[0];
  for (int i = t; i < LL; i += 256)
    pr[(size_t)row * LL + i] = (_Float16)(__expf(s[i] - mx) * inv);
}

// ---- 7) c = probs1 @ v  (K=L), stored transposed cT [B,D,P] --------------
__global__ __launch_bounds__(128) void k_c(const _Float16* __restrict__ pr,
                                           const _Float16* __restrict__ vtb,
                                           _Float16* __restrict__ ctb) {
  int lane = threadIdx.x & 31, wave = threadIdx.x >> 5;
  int m0 = blockIdx.x * 32;           // over B*P
  int b = m0 >> 7;
  int n0 = blockIdx.y * 256 + wave * 64;
  int ml = lane & 15, h = lane >> 4;
  const _Float16* arow0 = pr + (size_t)(m0 + ml) * LL;
  const _Float16* arow1 = pr + (size_t)(m0 + 16 + ml) * LL;
  v8f z = {0.f, 0.f, 0.f, 0.f, 0.f, 0.f, 0.f, 0.f};
  v8f acc[2][4] = {{z, z, z, z}, {z, z, z, z}};
  for (int k0 = 0; k0 < LL; k0 += 32) {
    v16h a0 = load_afrag(arow0 + k0, h);
    v16h a1 = load_afrag(arow1 + k0, h);
#pragma unroll
    for (int f = 0; f < 4; ++f) {
      v16h bf = load_bfrag(vtb + ((size_t)b * DD + n0 + f * 16 + ml) * LL + k0, h);
      acc[0][f] = wmma_f16(a0, bf, acc[0][f]);
      acc[1][f] = wmma_f16(a1, bf, acc[1][f]);
    }
  }
#pragma unroll
  for (int f = 0; f < 4; ++f) {
    int n = n0 + f * 16 + ml;
#pragma unroll
    for (int t = 0; t < 2; ++t)
#pragma unroll
      for (int r = 0; r < 8; ++r) {
        int p = (m0 & (PP - 1)) + 16 * t + r + 8 * h;
        ctb[((size_t)b * DD + n) * PP + p] = (_Float16)acc[t][f][r];
      }
  }
}

// ---- 8) stage-2 scores: S2[bL+l, p] = q . a  (K=D, N=P=128) --------------
__global__ __launch_bounds__(128) void k_s2(const _Float16* __restrict__ qb,
                                            const _Float16* __restrict__ ab,
                                            float* __restrict__ S) {
  int lane = threadIdx.x & 31, wave = threadIdx.x >> 5;
  int m0 = blockIdx.x * 32;           // over B*L
  int b = m0 >> 12;
  int n0 = wave * 32;                 // 4 waves cover all 128 agents
  int ml = lane & 15, h = lane >> 4;
  const _Float16* arow0 = qb + (size_t)(m0 + ml) * DD;
  const _Float16* arow1 = qb + (size_t)(m0 + 16 + ml) * DD;
  v8f z = {0.f, 0.f, 0.f, 0.f, 0.f, 0.f, 0.f, 0.f};
  v8f acc[2][2] = {{z, z}, {z, z}};
  for (int k0 = 0; k0 < DD; k0 += 32) {
    v16h a0 = load_afrag(arow0 + k0, h);
    v16h a1 = load_afrag(arow1 + k0, h);
#pragma unroll
    for (int f = 0; f < 2; ++f) {
      v16h bf = load_bfrag(ab + ((size_t)b * PP + n0 + f * 16 + ml) * DD + k0, h);
      acc[0][f] = wmma_f16(a0, bf, acc[0][f]);
      acc[1][f] = wmma_f16(a1, bf, acc[1][f]);
    }
  }
#pragma unroll
  for (int f = 0; f < 2; ++f) {
    int n = n0 + f * 16 + ml;
#pragma unroll
    for (int t = 0; t < 2; ++t)
#pragma unroll
      for (int r = 0; r < 8; ++r)
        S[(size_t)(m0 + 16 * t + r + 8 * h) * PP + n] = acc[t][f][r];
  }
}

// ---- 9) softmax over p (width 128), one wave per row, ds_swizzle reduce --
__global__ __launch_bounds__(128) void k_softmax2(const float* __restrict__ S,
                                                  _Float16* __restrict__ pr) {
  int wave = threadIdx.x >> 5, lane = threadIdx.x & 31;
  int row = blockIdx.x * 4 + wave;
  const float* s = S + (size_t)row * PP;
  float v[4], mx = -1e30f;
#pragma unroll
  for (int i = 0; i < 4; ++i) { v[i] = s[lane + 32 * i]; mx = fmaxf(mx, v[i]); }
  mx = wave_max(mx);
  float e[4], sm = 0.f;
#pragma unroll
  for (int i = 0; i < 4; ++i) { e[i] = __expf(v[i] - mx); sm += e[i]; }
  sm = wave_sum(sm);
  float inv = 1.0f / sm;
#pragma unroll
  for (int i = 0; i < 4; ++i)
    pr[(size_t)row * PP + lane + 32 * i] = (_Float16)(e[i] * inv);
}

// ---- 10) r = probs2 @ c (K=P=128), stored transposed into out [B,D,L] ----
__global__ __launch_bounds__(128) void k_r(const _Float16* __restrict__ pr,
                                           const _Float16* __restrict__ ctb,
                                           float* __restrict__ out) {
  int lane = threadIdx.x & 31, wave = threadIdx.x >> 5;
  int m0 = blockIdx.x * 32;           // over B*L
  int b = m0 >> 12, l0 = m0 & (LL - 1);
  int n0 = blockIdx.y * 256 + wave * 64;
  int ml = lane & 15, h = lane >> 4;
  const _Float16* arow0 = pr + (size_t)(m0 + ml) * PP;
  const _Float16* arow1 = pr + (size_t)(m0 + 16 + ml) * PP;
  v8f z = {0.f, 0.f, 0.f, 0.f, 0.f, 0.f, 0.f, 0.f};
  v8f acc[2][4] = {{z, z, z, z}, {z, z, z, z}};
#pragma unroll
  for (int k0 = 0; k0 < PP; k0 += 32) {
    v16h a0 = load_afrag(arow0 + k0, h);
    v16h a1 = load_afrag(arow1 + k0, h);
#pragma unroll
    for (int f = 0; f < 4; ++f) {
      v16h bf = load_bfrag(ctb + ((size_t)b * DD + n0 + f * 16 + ml) * PP + k0, h);
      acc[0][f] = wmma_f16(a0, bf, acc[0][f]);
      acc[1][f] = wmma_f16(a1, bf, acc[1][f]);
    }
  }
#pragma unroll
  for (int f = 0; f < 4; ++f) {
    int n = n0 + f * 16 + ml;
#pragma unroll
    for (int t = 0; t < 2; ++t)
#pragma unroll
      for (int r = 0; r < 8; ++r)
        out[((size_t)b * DD + n) * LL + (l0 + 16 * t + r + 8 * h)] = acc[t][f][r];
  }
}

// ---------------------------------------------------------------------------
extern "C" void kernel_launch(void* const* d_in, const int* in_sizes, int n_in,
                              void* d_out, int out_size, void* d_ws, size_t ws_size,
                              hipStream_t stream) {
  const float* x  = (const float*)d_in[0];
  const float* Wq = (const float*)d_in[1];
  const float* bq = (const float*)d_in[2];
  const float* Wk = (const float*)d_in[3];
  const float* bk = (const float*)d_in[4];
  const float* Wv = (const float*)d_in[5];
  const float* bv = (const float*)d_in[6];
  float* out = (float*)d_out;

  // Workspace carve-up (all offsets 256B aligned). Total ~163 MB.
  char* ws = (char*)d_ws;
  const size_t szBLD = (size_t)BL * DD * sizeof(_Float16);      // 33.5 MB
  _Float16* xtb = (_Float16*)(ws);
  _Float16* qb  = (_Float16*)(ws + szBLD);
  _Float16* kb  = (_Float16*)(ws + 2 * szBLD);
  _Float16* vtb = (_Float16*)(ws + 3 * szBLD);
  char* cur = ws + 4 * szBLD;
  _Float16* w3  = (_Float16*)cur; cur += (size_t)3 * DD * DD * sizeof(_Float16);
  _Float16* ab  = (_Float16*)cur; cur += (size_t)BP * DD * sizeof(_Float16);
  _Float16* ctb = (_Float16*)cur; cur += (size_t)BB * DD * PP * sizeof(_Float16);
  float*    S   = (float*)cur;    cur += (size_t)BP * LL * sizeof(float); // == BL*PP
  _Float16* pr  = (_Float16*)cur; // BP*LL halves, reused for BL*PP probs2

  k_transpose_x<<<dim3(LL / 32, DD / 32, BB), dim3(32, 8), 0, stream>>>(x, xtb);
  k_convert_w<<<dim3(3 * DD * DD / 256), dim3(256), 0, stream>>>(Wq, Wk, Wv, w3);
  k_proj<<<dim3(BL / 32, DD / 256, 3), dim3(128), 0, stream>>>(xtb, w3, bq, bk, bv,
                                                               qb, kb, vtb);
  k_pool<<<dim3(BP), dim3(256), 0, stream>>>(qb, ab);
  k_s1<<<dim3(BP / 32, LL / 256), dim3(128), 0, stream>>>(ab, kb, S);
  k_softmax1<<<dim3(BP), dim3(256), 0, stream>>>(S, pr);
  k_c<<<dim3(BP / 32, DD / 256), dim3(128), 0, stream>>>(pr, vtb, ctb);
  k_s2<<<dim3(BL / 32), dim3(128), 0, stream>>>(qb, ab, S);
  k_softmax2<<<dim3(BL / 4), dim3(128), 0, stream>>>(S, pr);
  k_r<<<dim3(BL / 32, DD / 256), dim3(128), 0, stream>>>(pr, ctb, out);
}
